// MixtureOfExperts_9096740733493
// MI455X (gfx1250) — compile-verified, matrix operand order checked by reading, not run on hardware
//
#include <hip/hip_runtime.h>
#include <hip/hip_bf16.h>

#define NEXP   8
#define HID    2048
#define INTERD 2048
#define NTOK   8192
#define SLIMIT 7.0f

// GEMM tiling
#define MT  128
#define NT  64
#define KT  32
#define LDA 40   // KT + 8 pad (bf16 elems); 80B row stride, 16B aligned
#define LDB 40

typedef __bf16 bf16;
typedef bf16  v16bf  __attribute__((ext_vector_type(16)));
typedef bf16  bf16x8 __attribute__((ext_vector_type(8)));
typedef bf16  bf16x4 __attribute__((ext_vector_type(4)));
typedef float v8f    __attribute__((ext_vector_type(8)));

// ---------------------------------------------------------------- router ----
__global__ __launch_bounds__(256) void moe_router(
    const float* __restrict__ X, const float* __restrict__ Wr,
    int* __restrict__ sel, float* __restrict__ prob)
{
  const int tid  = threadIdx.x;
  const int lane = tid & 31;
  const int t    = blockIdx.x * 8 + (tid >> 5);   // one wave per token

  float acc[NEXP];
  #pragma unroll
  for (int e = 0; e < NEXP; ++e) acc[e] = 0.f;

  const float* xr = X + (size_t)t * HID;
  for (int h = lane; h < HID; h += 32) {
    const float xv = xr[h];
    const float* wr = Wr + h * NEXP;
    #pragma unroll
    for (int e = 0; e < NEXP; ++e) acc[e] += xv * wr[e];
  }
  #pragma unroll
  for (int off = 16; off > 0; off >>= 1) {
    #pragma unroll
    for (int e = 0; e < NEXP; ++e) acc[e] += __shfl_xor(acc[e], off, 32);
  }
  if (lane == 0) {
    int b0 = 0; float v0 = acc[0];
    #pragma unroll
    for (int e = 1; e < NEXP; ++e) if (acc[e] > v0) { v0 = acc[e]; b0 = e; }
    int b1 = (b0 == 0) ? 1 : 0; float v1 = acc[b1];
    #pragma unroll
    for (int e = 0; e < NEXP; ++e) if (e != b0 && acc[e] > v1) { v1 = acc[e]; b1 = e; }
    const float e1  = __expf(v1 - v0);      // softmax over the two logits
    const float inv = 1.f / (1.f + e1);
    sel[t * 2 + 0]  = b0;  sel[t * 2 + 1]  = b1;
    prob[t * 2 + 0] = inv; prob[t * 2 + 1] = e1 * inv;
  }
}

// ------------------------------------------------- deterministic compaction -
__global__ __launch_bounds__(256) void moe_assign(
    const int* __restrict__ sel, const float* __restrict__ prob,
    int* __restrict__ counts, int* __restrict__ offsets,
    int* __restrict__ tok_list, float* __restrict__ gate_list)
{
  const int tid = threadIdx.x, lane = tid & 31, e = tid >> 5; // wave e -> expert e
  int cnt = 0;
  for (int t0 = 0; t0 < NTOK; t0 += 32) {
    const int t  = t0 + lane;
    const int s0 = sel[t * 2 + 0];
    const int s1 = sel[t * 2 + 1];
    const bool m0 = (s0 == e), m1 = (s1 == e);
    const bool m  = m0 | m1;
    const unsigned long long bal = __ballot(m);
    const int pos = cnt + (int)__popcll(bal & ((1ull << lane) - 1ull));
    if (m) {
      tok_list[e * NTOK + pos]  = t;
      gate_list[e * NTOK + pos] = m0 ? prob[t * 2] : prob[t * 2 + 1];
    }
    cnt += (int)__popcll(bal);
  }
  if (lane == 0) counts[e] = cnt;
  __syncthreads();
  if (tid == 0) {
    int s = 0;
    for (int i = 0; i < NEXP; ++i) { offsets[i] = s; s += counts[i]; }
    offsets[NEXP] = s;
  }
}

// ------------------------------------------------------------------ memset --
__global__ __launch_bounds__(256) void moe_zero(float* __restrict__ p, int n4)
{
  const float4 z = make_float4(0.f, 0.f, 0.f, 0.f);
  for (int i = blockIdx.x * blockDim.x + threadIdx.x; i < n4;
       i += gridDim.x * blockDim.x)
    ((float4*)p)[i] = z;
}

// -------------------------------------------------- shared frag load helper -
__device__ __forceinline__ v16bf load_frag(const bf16* base, int half)
{
  // A/B 16-bit frag (ISA 7.12.2): lane holds K = {8h..8h+7, 16+8h..16+8h+7}
  const bf16x8 s0 = *(const bf16x8*)(base + 8 * half);
  const bf16x8 s1 = *(const bf16x8*)(base + 16 + 8 * half);
  v16bf a;
  #pragma unroll
  for (int j = 0; j < 8; ++j) { a[j] = s0[j]; a[8 + j] = s1[j]; }
  return a;
}

// ------------------------------------------------- GEMM1: fused gate & up ---
__global__ __launch_bounds__(256) void moe_gemm1(
    const float* __restrict__ X, const float* __restrict__ Wg,
    const float* __restrict__ Wu,
    const int* __restrict__ counts, const int* __restrict__ offsets,
    const int* __restrict__ tok_list, bf16* __restrict__ H)
{
  const int e     = blockIdx.z;
  const int count = counts[e];
  const int mbase = blockIdx.x * MT;
  if (mbase >= count) return;                 // block-uniform; EXEC stays full
  const int nbase = blockIdx.y * NT;
  const int tid   = threadIdx.x;
  const int lane  = tid & 31;
  const int wave  = tid >> 5;
  const int wm    = (wave & 3) * 32;          // 4 waves in M
  const int wn    = (wave >> 2) * 32;         // 2 waves in N
  const int half  = lane >> 4;
  const int lrow  = lane & 15;

  const int*   tl = tok_list + e * NTOK;
  const float* wg = Wg + (size_t)e * HID * INTERD;
  const float* wu = Wu + (size_t)e * HID * INTERD;

  __shared__ bf16 lA [MT * LDA];
  __shared__ bf16 lBg[NT * LDB];
  __shared__ bf16 lBu[NT * LDB];

  // ---- hoist the token gather out of the K loop: per-chunk base pointers ----
  const float* srcA[4];
  bool         okA[4];
  int          rA[4], kA[4];
  #pragma unroll
  for (int i = 0; i < 4; ++i) {
    const int linear = tid + i * 256;         // 0..1023
    rA[i] = linear >> 3;                      // 0..127
    kA[i] = (linear & 7) << 2;                // 0,4,...,28
    const int row  = mbase + rA[i];
    okA[i] = (row < count);
    const int rowc = okA[i] ? row : 0;        // count > 0: always in-bounds
    srcA[i] = X + (size_t)tl[rowc] * HID + kA[i];
  }
  // B-tile per-chunk base pointers (K-invariant part)
  const float* srcBg[2];
  const float* srcBu[2];
  int nB[2], kB[2];
  #pragma unroll
  for (int i = 0; i < 2; ++i) {
    const int linear = tid + i * 256;         // 0..511
    kB[i] = linear >> 4;                      // 0..31
    nB[i] = (linear & 15) << 2;               // 0..60
    srcBg[i] = wg + (size_t)kB[i] * INTERD + nbase + nB[i];
    srcBu[i] = wu + (size_t)kB[i] * INTERD + nbase + nB[i];
  }

  v8f accg[2][2], accu[2][2];
  #pragma unroll
  for (int i = 0; i < 2; ++i)
    #pragma unroll
    for (int j = 0; j < 2; ++j) {
      #pragma unroll
      for (int r = 0; r < 8; ++r) { accg[i][j][r] = 0.f; accu[i][j][r] = 0.f; }
    }

  for (int k0 = 0; k0 < HID; k0 += KT) {
    // ---- issue all global loads for this step first (latency overlap) ----
    float4 va[4], vg[2], vu[2];
    #pragma unroll
    for (int i = 0; i < 4; ++i) va[i] = *(const float4*)(srcA[i] + k0);
    #pragma unroll
    for (int i = 0; i < 2; ++i) {
      vg[i] = *(const float4*)(srcBg[i] + (size_t)k0 * INTERD);
      vu[i] = *(const float4*)(srcBu[i] + (size_t)k0 * INTERD);
    }
    if (k0 + KT < HID) {    // prefetch next weight tiles (dominant stream)
      __builtin_prefetch(srcBg[0] + (size_t)(k0 + KT) * INTERD, 0, 3);
      __builtin_prefetch(srcBg[1] + (size_t)(k0 + KT) * INTERD, 0, 3);
      __builtin_prefetch(srcBu[0] + (size_t)(k0 + KT) * INTERD, 0, 3);
      __builtin_prefetch(srcBu[1] + (size_t)(k0 + KT) * INTERD, 0, 3);
    }

    __syncthreads();        // previous iteration's frag reads done
    // ---- A tile: 128 x 32 f32 -> bf16 (branchless zero for OOB rows) ----
    #pragma unroll
    for (int i = 0; i < 4; ++i) {
      float4 v = va[i];
      if (!okA[i]) v = make_float4(0.f, 0.f, 0.f, 0.f);   // v_cndmask, no branch
      const bf16x4 pk = { (bf16)v.x, (bf16)v.y, (bf16)v.z, (bf16)v.w };
      *(bf16x4*)(&lA[rA[i] * LDA + kA[i]]) = pk;
    }
    // ---- B tiles: transposed to [n][k] in LDS ----
    #pragma unroll
    for (int i = 0; i < 2; ++i) {
      const int kk = kB[i], n4 = nB[i];
      lBg[(n4 + 0) * LDB + kk] = (bf16)vg[i].x;
      lBg[(n4 + 1) * LDB + kk] = (bf16)vg[i].y;
      lBg[(n4 + 2) * LDB + kk] = (bf16)vg[i].z;
      lBg[(n4 + 3) * LDB + kk] = (bf16)vg[i].w;
      lBu[(n4 + 0) * LDB + kk] = (bf16)vu[i].x;
      lBu[(n4 + 1) * LDB + kk] = (bf16)vu[i].y;
      lBu[(n4 + 2) * LDB + kk] = (bf16)vu[i].z;
      lBu[(n4 + 3) * LDB + kk] = (bf16)vu[i].w;
    }
    __syncthreads();

    v16bf a[2], b[2];
    #pragma unroll
    for (int tm = 0; tm < 2; ++tm)
      a[tm] = load_frag(&lA[(wm + tm * 16 + lrow) * LDA], half);

    #pragma unroll
    for (int tn = 0; tn < 2; ++tn)
      b[tn] = load_frag(&lBg[(wn + tn * 16 + lrow) * LDB], half);
    #pragma unroll
    for (int tm = 0; tm < 2; ++tm)
      #pragma unroll
      for (int tn = 0; tn < 2; ++tn)
        accg[tm][tn] = __builtin_amdgcn_wmma_f32_16x16x32_bf16(
            false, a[tm], false, b[tn], (short)0, accg[tm][tn], false, false);

    #pragma unroll
    for (int tn = 0; tn < 2; ++tn)
      b[tn] = load_frag(&lBu[(wn + tn * 16 + lrow) * LDB], half);
    #pragma unroll
    for (int tm = 0; tm < 2; ++tm)
      #pragma unroll
      for (int tn = 0; tn < 2; ++tn)
        accu[tm][tn] = __builtin_amdgcn_wmma_f32_16x16x32_bf16(
            false, a[tm], false, b[tn], (short)0, accu[tm][tn], false, false);
  }

  // ---- epilogue: h = clip(silu(g)) * u  -> bf16 scratch H ----
  const int hoff = offsets[e];
  #pragma unroll
  for (int tm = 0; tm < 2; ++tm) {
    #pragma unroll
    for (int r = 0; r < 8; ++r) {
      const int rowg = mbase + wm + tm * 16 + r + 8 * half;     // C/D: M = r + 8*half
      if (rowg < count) {
        bf16* hrow = H + (size_t)(hoff + rowg) * INTERD + nbase + wn + lrow;
        #pragma unroll
        for (int tn = 0; tn < 2; ++tn) {
          const float g = accg[tm][tn][r];
          const float u = accu[tm][tn][r];
          float s = g / (1.f + __expf(-g));
          s = fminf(fmaxf(s, -SLIMIT), SLIMIT);
          hrow[tn * 16] = (bf16)(s * u);                         // C/D: N = lane%16
        }
      }
    }
  }
}

// ------------------------------------------ GEMM2: down proj + scatter-add --
__global__ __launch_bounds__(256) void moe_gemm2(
    const bf16* __restrict__ H, const float* __restrict__ Wd,
    const int* __restrict__ counts, const int* __restrict__ offsets,
    const int* __restrict__ tok_list, const float* __restrict__ gate_list,
    float* __restrict__ out)
{
  const int e     = blockIdx.z;
  const int count = counts[e];
  const int mbase = blockIdx.x * MT;
  if (mbase >= count) return;
  const int nbase = blockIdx.y * NT;
  const int tid   = threadIdx.x;
  const int lane  = tid & 31;
  const int wave  = tid >> 5;
  const int wm    = (wave & 3) * 32;
  const int wn    = (wave >> 2) * 32;
  const int half  = lane >> 4;
  const int lrow  = lane & 15;

  const float* wd = Wd + (size_t)e * INTERD * HID;
  const int  hoff = offsets[e];

  __shared__ bf16 lA[MT * LDA];
  __shared__ bf16 lB[NT * LDB];

  // hoisted A-tile (H rows, already bf16) base pointers
  const bf16* srcA[2];
  bool        okA[2];
  int         rA[2], kA[2];
  #pragma unroll
  for (int i = 0; i < 2; ++i) {
    const int linear = tid + i * 256;       // 0..511
    rA[i] = linear >> 2;                    // 0..127
    kA[i] = (linear & 3) << 3;              // 0,8,16,24
    const int row  = mbase + rA[i];
    okA[i] = (row < count);
    const int rowc = okA[i] ? row : 0;
    srcA[i] = H + (size_t)(hoff + rowc) * INTERD + kA[i];
  }
  const float* srcB[2];
  int nB[2], kB[2];
  #pragma unroll
  for (int i = 0; i < 2; ++i) {
    const int linear = tid + i * 256;
    kB[i] = linear >> 4;
    nB[i] = (linear & 15) << 2;
    srcB[i] = wd + (size_t)kB[i] * HID + nbase + nB[i];
  }

  v8f acc[2][2];
  #pragma unroll
  for (int i = 0; i < 2; ++i)
    #pragma unroll
    for (int j = 0; j < 2; ++j) {
      #pragma unroll
      for (int r = 0; r < 8; ++r) acc[i][j][r] = 0.f;
    }

  for (int k0 = 0; k0 < INTERD; k0 += KT) {
    bf16x8 va[2];
    float4 vb[2];
    #pragma unroll
    for (int i = 0; i < 2; ++i) va[i] = *(const bf16x8*)(srcA[i] + k0);
    #pragma unroll
    for (int i = 0; i < 2; ++i) vb[i] = *(const float4*)(srcB[i] + (size_t)k0 * HID);
    if (k0 + KT < INTERD) {
      __builtin_prefetch(srcB[0] + (size_t)(k0 + KT) * HID, 0, 3);
      __builtin_prefetch(srcB[1] + (size_t)(k0 + KT) * HID, 0, 3);
    }

    __syncthreads();
    #pragma unroll
    for (int i = 0; i < 2; ++i) {
      bf16x8 v = va[i];
      if (!okA[i]) {
        #pragma unroll
        for (int j = 0; j < 8; ++j) v[j] = (bf16)0.f;
      }
      *(bf16x8*)(&lA[rA[i] * LDA + kA[i]]) = v;
    }
    #pragma unroll
    for (int i = 0; i < 2; ++i) {
      const int kk = kB[i], n4 = nB[i];
      lB[(n4 + 0) * LDB + kk] = (bf16)vb[i].x;
      lB[(n4 + 1) * LDB + kk] = (bf16)vb[i].y;
      lB[(n4 + 2) * LDB + kk] = (bf16)vb[i].z;
      lB[(n4 + 3) * LDB + kk] = (bf16)vb[i].w;
    }
    __syncthreads();

    v16bf a[2], b[2];
    #pragma unroll
    for (int tm = 0; tm < 2; ++tm)
      a[tm] = load_frag(&lA[(wm + tm * 16 + lrow) * LDA], half);
    #pragma unroll
    for (int tn = 0; tn < 2; ++tn)
      b[tn] = load_frag(&lB[(wn + tn * 16 + lrow) * LDB], half);
    #pragma unroll
    for (int tm = 0; tm < 2; ++tm)
      #pragma unroll
      for (int tn = 0; tn < 2; ++tn)
        acc[tm][tn] = __builtin_amdgcn_wmma_f32_16x16x32_bf16(
            false, a[tm], false, b[tn], (short)0, acc[tm][tn], false, false);
  }

  // ---- epilogue: out[token] += gate * acc  (exactly 2 experts per token) ----
  #pragma unroll
  for (int tm = 0; tm < 2; ++tm) {
    #pragma unroll
    for (int r = 0; r < 8; ++r) {
      const int rowg = mbase + wm + tm * 16 + r + 8 * half;
      if (rowg < count) {
        const int tok = tok_list[e * NTOK + rowg];
        const float w = gate_list[e * NTOK + rowg];
        float* orow = out + (size_t)tok * HID + nbase + wn + lrow;
        #pragma unroll
        for (int tn = 0; tn < 2; ++tn)
          unsafeAtomicAdd(&orow[tn * 16], w * acc[tm][tn][r]);
      }
    }
  }
}

// ----------------------------------------------------------------- launch ---
extern "C" void kernel_launch(void* const* d_in, const int* in_sizes, int n_in,
                              void* d_out, int out_size, void* d_ws, size_t ws_size,
                              hipStream_t stream)
{
  const float* X  = (const float*)d_in[0];
  const float* Wr = (const float*)d_in[1];
  const float* Wg = (const float*)d_in[2];
  const float* Wu = (const float*)d_in[3];
  const float* Wd = (const float*)d_in[4];
  float* out = (float*)d_out;

  char* ws = (char*)d_ws;
  int*   counts    = (int*)  (ws + 0);                        // 32 B
  int*   offsets   = (int*)  (ws + 256);                      // 36 B
  int*   sel       = (int*)  (ws + 512);                      // 64 KB
  float* prob      = (float*)(ws + 512 + 65536);              // 64 KB
  int*   tok_list  = (int*)  (ws + 512 + 131072);             // 256 KB
  float* gate_list = (float*)(ws + 512 + 131072 + 262144);    // 256 KB
  bf16*  H         = (bf16*) (ws + 512 + 131072 + 524288);    // 64 MB (16384x2048 bf16)

  moe_router<<<NTOK / 8, 256, 0, stream>>>(X, Wr, sel, prob);
  moe_assign<<<1, 256, 0, stream>>>(sel, prob, counts, offsets, tok_list, gate_list);
  moe_zero<<<2048, 256, 0, stream>>>(out, (NTOK * HID) / 4);

  dim3 g1(NTOK / MT, INTERD / NT, NEXP);   // 64 x 32 x 8, early-exit on count
  moe_gemm1<<<g1, 256, 0, stream>>>(X, Wg, Wu, counts, offsets, tok_list, H);

  dim3 g2(NTOK / MT, HID / NT, NEXP);
  moe_gemm2<<<g2, 256, 0, stream>>>(H, Wd, counts, offsets, tok_list, gate_list, out);
}
